// MultiheadAttention_56092272886085
// MI455X (gfx1250) — matrix operand order repaired and checked by reading, not compiled, not run on hardware
//
#include <hip/hip_runtime.h>

// ---------------------------------------------------------------------------
// Types
// ---------------------------------------------------------------------------
typedef __bf16 bf16_t;
typedef __attribute__((ext_vector_type(16))) __bf16 v16bf;
typedef __attribute__((ext_vector_type(4)))  __bf16 v4bf;
typedef __attribute__((ext_vector_type(8)))  float  v8f;

union FragU { v16bf v; uint4 u[2]; };

#define LOG2E 1.4426950408889634f

// ---------------------------------------------------------------------------
// WMMA wrapper: D = A(16x32 bf16) x B(32x16 bf16) + C(16x16 f32)
// ---------------------------------------------------------------------------
__device__ __forceinline__ v8f wmma_bf16(v16bf a, v16bf b, v8f c) {
  return __builtin_amdgcn_wmma_f32_16x16x32_bf16(
      /*neg_a=*/false, a, /*neg_b=*/false, b,
      /*c_mod=*/(short)0, c, /*reuse_a=*/false, /*reuse_b=*/false);
}

// A fragment: 16(M) x 32(K) bf16 tile, row-major source, stride in elements.
// Lane m holds row m&15; lanes 0-15: K {0..7,16..23}; lanes 16-31: K {8..15,24..31}.
__device__ __forceinline__ v16bf load_a_frag(const bf16_t* base, int stride, int lane) {
  const bf16_t* row = base + (lane & 15) * stride + ((lane & 16) ? 8 : 0);
  FragU f;
  f.u[0] = *(const uint4*)(row);
  f.u[1] = *(const uint4*)(row + 16);
  return f.v;
}

// B fragment: 32(K) x 16(N) bf16; source is "N-major": baseT[n*stride + k].
// Lane n holds column n&15; lanes 0-15: K {0..7,16..23}; lanes 16-31: K {8..15,24..31}.
__device__ __forceinline__ v16bf load_b_frag(const bf16_t* baseT, int stride, int lane) {
  const bf16_t* row = baseT + (lane & 15) * stride + ((lane & 16) ? 8 : 0);
  FragU f;
  f.u[0] = *(const uint4*)(row);
  f.u[1] = *(const uint4*)(row + 16);
  return f.v;
}

// 16-lane-group reductions (wave32: masks 1,2,4,8 stay within each half)
__device__ __forceinline__ float rmax16(float v) {
  v = fmaxf(v, __shfl_xor(v, 1, 32));
  v = fmaxf(v, __shfl_xor(v, 2, 32));
  v = fmaxf(v, __shfl_xor(v, 4, 32));
  v = fmaxf(v, __shfl_xor(v, 8, 32));
  return v;
}
__device__ __forceinline__ float rsum16(float v) {
  v += __shfl_xor(v, 1, 32);
  v += __shfl_xor(v, 2, 32);
  v += __shfl_xor(v, 4, 32);
  v += __shfl_xor(v, 8, 32);
  return v;
}

// ---------------------------------------------------------------------------
// f32 -> bf16 convert, 4 elements/thread (b128 load / b64 store)
// ---------------------------------------------------------------------------
__global__ void convert_f32_bf16_kernel(const float* __restrict__ in,
                                        bf16_t* __restrict__ out, int n4) {
  int i = blockIdx.x * blockDim.x + threadIdx.x;
  if (i < n4) {
    float4 f = ((const float4*)in)[i];
    v4bf o;
    o.x = (bf16_t)f.x; o.y = (bf16_t)f.y; o.z = (bf16_t)f.z; o.w = (bf16_t)f.w;
    ((v4bf*)out)[i] = o;
  }
}

// in[rows][cols] f32 -> out[cols][rows] bf16; LDS-tiled 32x32, both sides coalesced.
// Grid: (cols/32, rows/32), block (32, 8). rows, cols multiples of 32.
__global__ void transpose_f32_bf16_kernel(const float* __restrict__ in,
                                          bf16_t* __restrict__ out,
                                          int rows, int cols) {
  __shared__ float tile[32][33];
  int c0 = blockIdx.x * 32;
  int r0 = blockIdx.y * 32;
  int tx = threadIdx.x, ty = threadIdx.y;
#pragma unroll
  for (int i = 0; i < 32; i += 8)
    tile[ty + i][tx] = in[(r0 + ty + i) * cols + c0 + tx];
  __syncthreads();
#pragma unroll
  for (int i = 0; i < 32; i += 8)
    out[(c0 + ty + i) * rows + r0 + tx] = (bf16_t)tile[tx][ty + i];
}

// ---------------------------------------------------------------------------
// Shared GEMM core: one wave computes a 64x64 f32 tile of A(MxK) x B^T(NxK)
// ---------------------------------------------------------------------------
__device__ __forceinline__ void gemm_wave_64x64(const bf16_t* __restrict__ A, int lda,
                                                const bf16_t* __restrict__ BT, int ldb,
                                                int K, v8f c[4][4], int lane) {
  for (int kb = 0; kb < K; kb += 32) {
    v16bf a[4], b[4];
#pragma unroll
    for (int i = 0; i < 4; ++i) a[i] = load_a_frag(A + i * 16 * lda + kb, lda, lane);
#pragma unroll
    for (int j = 0; j < 4; ++j) b[j] = load_b_frag(BT + j * 16 * ldb + kb, ldb, lane);
#pragma unroll
    for (int i = 0; i < 4; ++i)
#pragma unroll
      for (int j = 0; j < 4; ++j)
        c[i][j] = wmma_bf16(a[i], b[j], c[i][j]);
  }
}

// ---------------------------------------------------------------------------
// QKV projection: [4096,1024] x [1024,3072] + b_attn
// part (q/k/v) is uniform per WORKGROUP: WG N-tile = 256 and 256 | 1024.
// h is uniform per wave (64 | wave N-tile); b is uniform per wave.
// Writes: Qbf,Kbf  bf16 [B,H,S,64];  Vt bf16 [B,H,64,S];
//         presentK/presentV f32 [B,H,S,64]  (reference "present")
// Grid: (32, 12) x 256 threads
// ---------------------------------------------------------------------------
__global__ __launch_bounds__(256) void qkv_gemm_kernel(
    const bf16_t* __restrict__ X, const bf16_t* __restrict__ WtA,
    const float* __restrict__ b_attn,
    bf16_t* __restrict__ Qbf, bf16_t* __restrict__ Kbf, bf16_t* __restrict__ Vt,
    float* __restrict__ presentK, float* __restrict__ presentV) {
  int lane = threadIdx.x & 31;
  int wave = threadIdx.x >> 5;
  int m0 = blockIdx.x * 128 + (wave >> 2) * 64;
  int n0 = blockIdx.y * 256 + (wave & 3) * 64;

  v8f zero = {};
  v8f c[4][4];
#pragma unroll
  for (int i = 0; i < 4; ++i)
#pragma unroll
    for (int j = 0; j < 4; ++j) c[i][j] = zero;

  gemm_wave_64x64(X + m0 * 1024, 1024, WtA + n0 * 1024, 1024, 1024, c, lane);

  int colb = lane & 15;
  int rowhalf = (lane & 16) ? 8 : 0;

  // Hoisted bias (4 distinct columns per lane, reused across 32 elements)
  float bias[4];
#pragma unroll
  for (int j = 0; j < 4; ++j) bias[j] = b_attn[n0 + j * 16 + colb];

  int part = n0 >> 10;                 // uniform per workgroup: 0=q 1=k 2=v
  int b  = m0 >> 11;                   // uniform per wave (64 | 2048)
  int s0 = m0 & 2047;
  int h  = (n0 & 1023) >> 6;           // uniform per wave (64-wide N tile)
  int bh = b * 16 + h;

  if (part == 0) {
    bf16_t* __restrict__ qb = Qbf + bh * 2048 * 64;
#pragma unroll
    for (int i = 0; i < 4; ++i)
#pragma unroll
      for (int j = 0; j < 4; ++j)
#pragma unroll
        for (int jj = 0; jj < 8; ++jj) {
          int s  = s0 + i * 16 + rowhalf + jj;
          int dd = j * 16 + colb;
          qb[s * 64 + dd] = (bf16_t)(c[i][j][jj] + bias[j]);
        }
  } else if (part == 1) {
    bf16_t* __restrict__ kb = Kbf + bh * 2048 * 64;
    float*  __restrict__ pk = presentK + bh * 2048 * 64;
#pragma unroll
    for (int i = 0; i < 4; ++i)
#pragma unroll
      for (int j = 0; j < 4; ++j)
#pragma unroll
        for (int jj = 0; jj < 8; ++jj) {
          int s  = s0 + i * 16 + rowhalf + jj;
          int dd = j * 16 + colb;
          float val = c[i][j][jj] + bias[j];
          kb[s * 64 + dd] = (bf16_t)val;
          pk[s * 64 + dd] = val;
        }
  } else {
    bf16_t* __restrict__ vt = Vt + bh * 64 * 2048;
    float*  __restrict__ pv = presentV + bh * 2048 * 64;
#pragma unroll
    for (int i = 0; i < 4; ++i)
#pragma unroll
      for (int j = 0; j < 4; ++j)
#pragma unroll
        for (int jj = 0; jj < 8; ++jj) {
          int s  = s0 + i * 16 + rowhalf + jj;
          int dd = j * 16 + colb;
          float val = c[i][j][jj] + bias[j];
          vt[dd * 2048 + s] = (bf16_t)val;   // pre-transposed for PV B-operand
          pv[s * 64 + dd] = val;
        }
  }
}

// ---------------------------------------------------------------------------
// Output projection: out[4096,1024] = A x Wproj + b_proj (f32 out)
// Grid: (32, 4) x 256 threads
// ---------------------------------------------------------------------------
__global__ __launch_bounds__(256) void proj_gemm_kernel(
    const bf16_t* __restrict__ A, const bf16_t* __restrict__ WtP,
    const float* __restrict__ b_proj, float* __restrict__ out) {
  int lane = threadIdx.x & 31;
  int wave = threadIdx.x >> 5;
  int m0 = blockIdx.x * 128 + (wave >> 2) * 64;
  int n0 = blockIdx.y * 256 + (wave & 3) * 64;

  v8f zero = {};
  v8f c[4][4];
#pragma unroll
  for (int i = 0; i < 4; ++i)
#pragma unroll
    for (int j = 0; j < 4; ++j) c[i][j] = zero;

  gemm_wave_64x64(A + m0 * 1024, 1024, WtP + n0 * 1024, 1024, 1024, c, lane);

  int colb = lane & 15;
  int rowhalf = (lane & 16) ? 8 : 0;
  float bias[4];
#pragma unroll
  for (int j = 0; j < 4; ++j) bias[j] = b_proj[n0 + j * 16 + colb];

#pragma unroll
  for (int i = 0; i < 4; ++i)
#pragma unroll
    for (int j = 0; j < 4; ++j)
#pragma unroll
      for (int jj = 0; jj < 8; ++jj) {
        int row = m0 + i * 16 + rowhalf + jj;
        int col = n0 + j * 16 + colb;
        out[row * 1024 + col] = c[i][j][jj] + bias[j];
      }
}

// ---------------------------------------------------------------------------
// Causal flash attention. One wave = one (b,h) 16-query block.
// 32 keys/iter: 4 WMMAs (QK^T) + 4 WMMAs (PV). Mask path only where the
// tile touches the diagonal (wave-uniform branch).
// Grid: 512 x 256 threads (4096 waves total = B*H*S/16)
// ---------------------------------------------------------------------------
__global__ __launch_bounds__(256) void flash_attn_kernel(
    const bf16_t* __restrict__ Qbf, const bf16_t* __restrict__ Kbf,
    const bf16_t* __restrict__ Vt, bf16_t* __restrict__ Abf) {
  __shared__ __bf16 Plds[8 * 16 * 40];  // 8 waves x (16 rows x stride 40)

  int lane = threadIdx.x & 31;
  int wave = threadIdx.x >> 5;
  int wid = blockIdx.x * 8 + wave;     // 0..4095
  int bh = wid >> 7;                   // 0..31
  int qblk = wid & 127;
  int q0 = qblk * 16;
  int b = bh >> 4, h = bh & 15;

  // Q fragments are loop-invariant (16 queries x 64 d, two K=32 chunks)
  const bf16_t* Qrow = Qbf + (bh * 2048 + q0) * 64;
  v16bf aq0 = load_a_frag(Qrow, 64, lane);
  v16bf aq1 = load_a_frag(Qrow + 32, 64, lane);

  v8f zero = {};
  v8f o[4];
#pragma unroll
  for (int t = 0; t < 4; ++t) o[t] = zero;
  float mrow[8], lrow[8];
#pragma unroll
  for (int j = 0; j < 8; ++j) { mrow[j] = -1e30f; lrow[j] = 0.0f; }

  __bf16* Pw = &Plds[wave * 640];
  int colb = lane & 15;
  int rowhalf = (lane & 16) ? 8 : 0;
  const float scl = 1.0f / 64.0f;

  // Online-softmax tile update; `domask` is a call-site constant so the
  // compiler folds the masked compare/select out of the fast path.
  auto update_tile = [&](v8f s1, v8f s2, int kstart, bool domask) {
#pragma unroll
    for (int j = 0; j < 8; ++j) {
      int q = q0 + rowhalf + j;
      float v1 = s1[j] * scl;
      float v2 = s2[j] * scl;
      if (domask) {
        if (kstart + colb > q)      v1 = -1e10f;
        if (kstart + 16 + colb > q) v2 = -1e10f;
      }
      float mnew = fmaxf(mrow[j], rmax16(fmaxf(v1, v2)));
      float corr = __builtin_amdgcn_exp2f((mrow[j] - mnew) * LOG2E);
      float p1 = __builtin_amdgcn_exp2f((v1 - mnew) * LOG2E);
      float p2 = __builtin_amdgcn_exp2f((v2 - mnew) * LOG2E);
      lrow[j] = lrow[j] * corr + rsum16(p1 + p2);
      mrow[j] = mnew;
#pragma unroll
      for (int t = 0; t < 4; ++t) o[t][j] *= corr;
      int prow = rowhalf + j;
      Pw[prow * 40 + colb]      = (__bf16)p1;   // keys kstart..+15
      Pw[prow * 40 + 16 + colb] = (__bf16)p2;   // keys kstart+16..+31
    }
  };

  int kend = ((q0 + 16 + 31) >> 5) << 5;   // multiple of 32 covering the diagonal
  for (int kstart = 0; kstart < kend; kstart += 32) {
    // ---- S = Q K^T ----
    const bf16_t* Kt = Kbf + (bh * 2048 + kstart) * 64;
    v8f s1 = zero, s2 = zero;
    s1 = wmma_bf16(aq0, load_b_frag(Kt, 64, lane), s1);
    s1 = wmma_bf16(aq1, load_b_frag(Kt + 32, 64, lane), s1);
    s2 = wmma_bf16(aq0, load_b_frag(Kt + 16 * 64, 64, lane), s2);
    s2 = wmma_bf16(aq1, load_b_frag(Kt + 16 * 64 + 32, 64, lane), s2);

    // ---- scale, causal mask (only near diagonal), online softmax ----
    if (kstart + 31 > q0) update_tile(s1, s2, kstart, true);
    else                  update_tile(s1, s2, kstart, false);

    // per-wave LDS: hardware keeps DS ops in order; stop compiler reordering
    asm volatile("s_wait_dscnt 0" ::: "memory");

    // ---- O += P V  (P: 16x32 A-op from LDS; V^T rows are contiguous) ----
    v16bf ap = load_a_frag(Pw, 40, lane);
    const bf16_t* Vtt = Vt + (bh * 64) * 2048 + kstart;
#pragma unroll
    for (int t = 0; t < 4; ++t)
      o[t] = wmma_bf16(ap, load_b_frag(Vtt + (t * 16) * 2048, 2048, lane), o[t]);
  }

  // ---- normalize and write merged-heads activation (bf16 for proj GEMM) ----
#pragma unroll
  for (int j = 0; j < 8; ++j) {
    float inv = 1.0f / lrow[j];
    int q = q0 + rowhalf + j;
    int rowbase = (b * 2048 + q) * 1024 + h * 64;
#pragma unroll
    for (int t = 0; t < 4; ++t)
      Abf[rowbase + t * 16 + colb] = (bf16_t)(o[t][j] * inv);
  }
}

// ---------------------------------------------------------------------------
// Launch
// ---------------------------------------------------------------------------
extern "C" void kernel_launch(void* const* d_in, const int* in_sizes, int n_in,
                              void* d_out, int out_size, void* d_ws, size_t ws_size,
                              hipStream_t stream) {
  const float* x      = (const float*)d_in[0];  // [2,2048,1024]
  const float* w_attn = (const float*)d_in[1];  // [1024,3072]
  const float* b_attn = (const float*)d_in[2];  // [3072]
  const float* w_proj = (const float*)d_in[3];  // [1024,1024]
  const float* b_proj = (const float*)d_in[4];  // [1024]

  float* out      = (float*)d_out;              // [2,2048,1024]
  float* presentK = out + 4194304;              // present[0] = k [2,16,2048,64]
  float* presentV = out + 8388608;              // present[1] = v

  // Workspace partition (bf16): ~50 MB total
  bf16_t* w = (bf16_t*)d_ws;
  bf16_t* Xbf = w;  w += 4096 * 1024;
  bf16_t* WtA = w;  w += 3072 * 1024;   // w_attn^T, N-major
  bf16_t* WtP = w;  w += 1024 * 1024;   // w_proj^T, N-major
  bf16_t* Qbf = w;  w += 4194304;       // [B,H,S,64]
  bf16_t* Kbf = w;  w += 4194304;       // [B,H,S,64]
  bf16_t* Vt  = w;  w += 4194304;       // [B,H,64,S]
  bf16_t* Abf = w;  w += 4194304;       // merged heads [4096,1024]

  convert_f32_bf16_kernel<<<4096, 256, 0, stream>>>(x, Xbf, 1048576);
  transpose_f32_bf16_kernel<<<dim3(96, 32), dim3(32, 8), 0, stream>>>(w_attn, WtA, 1024, 3072);
  transpose_f32_bf16_kernel<<<dim3(32, 32), dim3(32, 8), 0, stream>>>(w_proj, WtP, 1024, 1024);

  qkv_gemm_kernel<<<dim3(32, 12), 256, 0, stream>>>(Xbf, WtA, b_attn,
                                                    Qbf, Kbf, Vt, presentK, presentV);
  flash_attn_kernel<<<512, 256, 0, stream>>>(Qbf, Kbf, Vt, Abf);
  proj_gemm_kernel<<<dim3(32, 4), 256, 0, stream>>>(Abf, WtP, b_proj, out);
}
